// GaussianUpsampling_63505386438804
// MI455X (gfx1250) — matrix operand order repaired and checked by reading, not compiled, not run on hardware
//
#include <hip/hip_runtime.h>
#include <hip/hip_bf16.h>

#define B_ 16
#define T_IN_ 512
#define D_ 256

typedef __attribute__((ext_vector_type(16))) _Float16 v16h;
typedef __attribute__((ext_vector_type(8)))  _Float16 v8h;
typedef __attribute__((ext_vector_type(8)))  float    v8f;

// ---------------------------------------------------------------------------
// Kernel A: enc[b][i][d] (f32) -> encT[b][d][i] (f16), tiled 32x32 transpose
// ---------------------------------------------------------------------------
__global__ __launch_bounds__(256) void enc_transpose_f16(
    const float* __restrict__ enc, _Float16* __restrict__ encT) {
  __shared__ _Float16 tile[32][33];
  const int b  = blockIdx.z;
  const int i0 = blockIdx.x * 32;
  const int d0 = blockIdx.y * 32;
  const int tx = threadIdx.x;      // 0..31
  const int ty = threadIdx.y;      // 0..7
  const float* e = enc + (size_t)b * T_IN_ * D_;
#pragma unroll
  for (int j = 0; j < 4; ++j) {
    int i = i0 + ty + 8 * j;
    tile[ty + 8 * j][tx] = (_Float16)e[(size_t)i * D_ + d0 + tx];
  }
  __syncthreads();
  _Float16* o = encT + (size_t)b * D_ * T_IN_;
#pragma unroll
  for (int j = 0; j < 4; ++j) {
    int d = d0 + ty + 8 * j;
    o[(size_t)d * T_IN_ + i0 + tx] = tile[tx][ty + 8 * j];
  }
}

// ---------------------------------------------------------------------------
// Kernel B: frame centers c[b][i] = cumsum(dur)[b][i] - 0.5*dur[b][i]
// ---------------------------------------------------------------------------
__global__ __launch_bounds__(512) void centers_kernel(
    const int* __restrict__ dur, float* __restrict__ c) {
  __shared__ float s[T_IN_];
  const int b = blockIdx.x;
  const int i = threadIdx.x;
  const float d = (float)dur[b * T_IN_ + i];
  s[i] = d;
  __syncthreads();
  for (int off = 1; off < T_IN_; off <<= 1) {
    float t = (i >= off) ? s[i - off] : 0.0f;
    __syncthreads();
    s[i] += t;
    __syncthreads();
  }
  c[b * T_IN_ + i] = s[i] - 0.5f * d;
}

// ---------------------------------------------------------------------------
// Kernel C: fused w computation + WMMA GEMM
// grid = (ceil(T_out/32), B), block = 256 threads (8 wave32)
// Block computes 32 t-rows; each wave owns a 32x32 output tile (2Mx2N WMMA
// tiles) so every A and every B fragment feeds 2 WMMAs.
// ---------------------------------------------------------------------------
__global__ __launch_bounds__(256) void gauss_upsample_fused(
    const float* __restrict__ c_all, const _Float16* __restrict__ encT,
    float* __restrict__ out, float* __restrict__ w_out, int T_out) {
  __shared__ float    w1s[T_IN_][33];   // padded: conflict-free row/col access
  __shared__ _Float16 wa[32][T_IN_];    // A-matrix tiles (t x i), f16
  __shared__ float    red[32][8];
  __shared__ float    inv_s[32];

  const int b   = blockIdx.y;
  const int t0  = blockIdx.x * 32;
  const int tid = threadIdx.x;

  // ---- pass 1: w1[i][t] = exp(-0.1*(t - c_i)^2), zero for t >= T_out ----
  {
    const float* cb = c_all + b * T_IN_;
    const float ca = cb[tid];
    const float cc = cb[tid + 256];
#pragma unroll
    for (int tl = 0; tl < 32; ++tl) {
      const bool  valid = (t0 + tl) < T_out;
      const float tv = (float)(t0 + tl + 1);
      const float da = tv - ca;
      const float db = tv - cc;
      w1s[tid][tl]       = valid ? __expf(-0.1f * da * da) : 0.0f;
      w1s[tid + 256][tl] = valid ? __expf(-0.1f * db * db) : 0.0f;
    }
  }
  __syncthreads();

  // ---- pass 2: column sums over i (512), then reciprocal with ==0 guard ----
  {
    const int tl = tid >> 3;     // 0..31
    const int r  = tid & 7;      // 0..7
    float s = 0.0f;
#pragma unroll 8
    for (int j = 0; j < 64; ++j) s += w1s[r + 8 * j][tl];
    red[tl][r] = s;
  }
  __syncthreads();
  if (tid < 32) {
    float s = 0.0f;
#pragma unroll
    for (int r = 0; r < 8; ++r) s += red[tid][r];
    inv_s[tid] = 1.0f / ((s == 0.0f) ? 1.0f : s);
  }
  __syncthreads();

  // ---- pass 3a: write w to HBM; each wave stores 128B contiguous runs ----
  // Streaming output: nontemporal so it does not evict encT from L2.
  {
    const int   tl    = tid & 31;
    const int   ibase = tid >> 5;   // 0..7
    const float inv   = inv_s[tl];
    const bool  valid = (t0 + tl) < T_out;
    float* wbase = w_out + (size_t)b * T_IN_ * T_out + (t0 + tl);
    for (int j = 0; j < 64; ++j) {
      const int i = ibase + 8 * j;
      const float w = w1s[i][tl] * inv;
      if (valid) __builtin_nontemporal_store(w, wbase + (size_t)i * T_out);
    }
  }
  // ---- pass 3b: normalized f16 A-tiles into LDS (i-major writes) ----
  {
#pragma unroll
    for (int tl = 0; tl < 32; ++tl) {
      const float inv = inv_s[tl];
      wa[tl][tid]       = (_Float16)(w1s[tid][tl] * inv);
      wa[tl][tid + 256] = (_Float16)(w1s[tid + 256][tl] * inv);
    }
  }
  __syncthreads();

  // ---- pass 4: out[t0:t0+32, :] = wa(32x512) * enc(512x256) via WMMA ----
  const int lane = tid & 31;
  const int wave = tid >> 5;
  const int row  = lane & 15;                 // A matrix M row (within tile)
  const int kbA  = (lane < 16) ? 0 : 16;      // A K-base per half-wave
  const int kbB  = (lane < 16) ? 0 : 16;      // B K-base per half-wave
  const int d0   = wave * 32;                 // each wave owns 32 d columns
  const _Float16* bt0 = encT + ((size_t)b * D_ + d0 + (lane & 15)) * T_IN_;
  const _Float16* bt1 = encT + ((size_t)b * D_ + d0 + 16 + (lane & 15)) * T_IN_;

  v8f acc00 = {}, acc01 = {}, acc10 = {}, acc11 = {};
#pragma unroll 2
  for (int k0 = 0; k0 < T_IN_; k0 += 32) {
    union { v16h v; v8h h[2]; } af0, af1;
    af0.h[0] = *(const v8h*)&wa[row][k0 + kbA];           // K = kbA + 0..7
    af0.h[1] = *(const v8h*)&wa[row][k0 + kbA + 8];       // K = kbA + 8..15
    af1.h[0] = *(const v8h*)&wa[row + 16][k0 + kbA];
    af1.h[1] = *(const v8h*)&wa[row + 16][k0 + kbA + 8];
    const v16h bv0 = *(const v16h*)(bt0 + k0 + kbB);      // K = kbB + 0..15
    const v16h bv1 = *(const v16h*)(bt1 + k0 + kbB);
    acc00 = __builtin_amdgcn_wmma_f32_16x16x32_f16(
        false, af0.v, false, bv0, (short)0, acc00, false, false);
    acc01 = __builtin_amdgcn_wmma_f32_16x16x32_f16(
        false, af0.v, false, bv1, (short)0, acc01, false, false);
    acc10 = __builtin_amdgcn_wmma_f32_16x16x32_f16(
        false, af1.v, false, bv0, (short)0, acc10, false, false);
    acc11 = __builtin_amdgcn_wmma_f32_16x16x32_f16(
        false, af1.v, false, bv1, (short)0, acc11, false, false);
  }

  // ---- store C/D tiles: VGPR r -> M = r + (lane<16?0:8), N = lane&15 ----
  {
    const int  Mb   = (lane < 16) ? 0 : 8;
    const int  N    = lane & 15;
    const bool full = (t0 + 32) <= T_out;     // uniform fast path
    if (full) {
#pragma unroll
      for (int r = 0; r < 8; ++r) {
        float* o0 = out + ((size_t)b * T_out + t0 + Mb + r) * D_;
        float* o1 = out + ((size_t)b * T_out + t0 + 16 + Mb + r) * D_;
        __builtin_nontemporal_store(acc00[r], o0 + d0 + N);
        __builtin_nontemporal_store(acc01[r], o0 + d0 + 16 + N);
        __builtin_nontemporal_store(acc10[r], o1 + d0 + N);
        __builtin_nontemporal_store(acc11[r], o1 + d0 + 16 + N);
      }
    } else {
#pragma unroll
      for (int r = 0; r < 8; ++r) {
        const int ta = t0 + Mb + r;
        const int tb = t0 + 16 + Mb + r;
        if (ta < T_out) {
          float* o0 = out + ((size_t)b * T_out + ta) * D_;
          __builtin_nontemporal_store(acc00[r], o0 + d0 + N);
          __builtin_nontemporal_store(acc01[r], o0 + d0 + 16 + N);
        }
        if (tb < T_out) {
          float* o1 = out + ((size_t)b * T_out + tb) * D_;
          __builtin_nontemporal_store(acc10[r], o1 + d0 + N);
          __builtin_nontemporal_store(acc11[r], o1 + d0 + 16 + N);
        }
      }
    }
  }
}

// ---------------------------------------------------------------------------
extern "C" void kernel_launch(void* const* d_in, const int* in_sizes, int n_in,
                              void* d_out, int out_size, void* d_ws, size_t ws_size,
                              hipStream_t stream) {
  (void)in_sizes; (void)n_in; (void)ws_size;
  const float* enc = (const float*)d_in[0];
  const int*   dur = (const int*)d_in[1];
  // d_in[2] = mask (unused by reference)

  const int T_out = out_size / (B_ * (T_IN_ + D_));  // out + w concatenated

  _Float16* encT = (_Float16*)d_ws;                              // 4 MB
  float*    cbuf = (float*)((char*)d_ws +
                            (size_t)B_ * D_ * T_IN_ * sizeof(_Float16));  // 32 KB
  float* out   = (float*)d_out;
  float* w_out = out + (size_t)B_ * T_out * D_;

  dim3 tgrid(T_IN_ / 32, D_ / 32, B_), tblk(32, 8);
  enc_transpose_f16<<<tgrid, tblk, 0, stream>>>(enc, encT);

  centers_kernel<<<B_, T_IN_, 0, stream>>>(dur, cbuf);

  dim3 fgrid((T_out + 31) / 32, B_), fblk(256);
  gauss_upsample_fused<<<fgrid, fblk, 0, stream>>>(cbuf, encT, out, w_out, T_out);
}
